// IRL_event_56461640073243
// MI455X (gfx1250) — matrix operand order repaired and checked by reading, not compiled, not run on hardware
//
#include <hip/hip_runtime.h>

// ---------------------------------------------------------------------------
// MI455X (gfx1250):  dx = -x + sigma(x) @ A^T + u*sigma(x);  out=[dx,-dx,0]
//
// GEMM = 17.2 GFLOP, HBM floor ~9us @ 23.3TB/s. bf16 hi/lo split x3 WMMA
// passes (fp32-class accuracy) on v_wmma_f32_16x16x32_bf16. Operand tiles are
// staged into LDS with GLOBAL_LOAD_ASYNC_TO_LDS_B128 (ASYNCcnt, coalesced
// global reads, swizzle-padded LDS rows -> conflict-free ds_load_b128
// fragment reads), double-buffered so the async DMA of chunk k+1 overlaps the
// 24 WMMAs of chunk k. Zero plane of the output is written coalesced in the
// elementwise pre-pass, not from the WMMA kernel.
// ---------------------------------------------------------------------------

#define BATCH 8192
#define DIM   1024

#define BM 128
#define BN 128
#define BK 32
#define ROWB  80                 // LDS row stride: 64B data + 16B pad (bank swizzle)
#define TILEB (128 * ROWB)       // 10240 B per staged tile
#define STAGEB (4 * TILEB)       // Sh,Sl,Ah,Al -> 40960 B per stage

typedef unsigned short u16;
typedef __attribute__((ext_vector_type(4)))  float f32x4;
typedef __attribute__((ext_vector_type(8)))  float v8f;
typedef __attribute__((ext_vector_type(4)))  u16   u16x4;
typedef __attribute__((ext_vector_type(8)))  u16   u16x8;
typedef __attribute__((ext_vector_type(16))) __bf16 v16bf;

union Frag { u16x8 h[2]; v16bf bf; };

__device__ __forceinline__ u16 f32_to_bf16(float f) {
    unsigned int u = __float_as_uint(f);
    u += 0x7FFFu + ((u >> 16) & 1u);          // round-to-nearest-even
    return (u16)(u >> 16);
}
__device__ __forceinline__ float bf16_to_f32(u16 h) {
    return __uint_as_float(((unsigned int)h) << 16);
}

// gfx1250 async DMA: LDS[vdst] = MEM[vaddr], tracked by ASYNCcnt (ISA 08 §4).
__device__ __forceinline__ void async_load16(unsigned lds_addr, const void* gptr) {
    asm volatile("global_load_async_to_lds_b128 %0, %1, off"
                 :: "v"(lds_addr), "v"((unsigned long long)(size_t)gptr)
                 : "memory");
}
__device__ __forceinline__ void wait_async0() {
    asm volatile("s_wait_asynccnt 0" ::: "memory");
}

// ---------------------------------------------------------------------------
// Pre-pass 1: elementwise over [B,D] (coalesced float4/thread)
//   sig -> Sh/Sl (bf16 hi/lo),  c0 = -x + u*sig -> C0 (f32)
//   Also writes the zero plane out[2*B*D ..] fully coalesced.
// ---------------------------------------------------------------------------
__global__ __launch_bounds__(256) void prep_state(
    const float* __restrict__ x, const float* __restrict__ ex,
    const float* __restrict__ W, const float* __restrict__ tgt,
    u16* __restrict__ Sh, u16* __restrict__ Sl, float* __restrict__ C0,
    float* __restrict__ out_zero)
{
    size_t i = (size_t)blockIdx.x * 256 + threadIdx.x;   // vec4 index
    f32x4 xv = ((const f32x4*)x)[i];
    f32x4 ev = ((const f32x4*)ex)[i];
    f32x4 wv = ((const f32x4*)W)[i];
    f32x4 tv = ((const f32x4*)tgt)[i & (DIM / 4 - 1)];

    u16x4 sh, sl;
    f32x4 c0;
#pragma unroll
    for (int c = 0; c < 4; ++c) {
        float xf   = xv[c];
        float tf   = tv[c];
        float bast = tf * tf / (1.0f + tf * tf);
        float u    = -(wv[c] * ((xf + ev[c]) - tf)) * bast;
        float sig  = xf * xf / (1.0f + xf * xf);
        u16   hi   = f32_to_bf16(sig);
        sh[c] = hi;
        sl[c] = f32_to_bf16(sig - bf16_to_f32(hi));
        c0[c] = -xf + u * sig;
    }
    ((u16x4*)Sh)[i] = sh;
    ((u16x4*)Sl)[i] = sl;
    ((f32x4*)C0)[i] = c0;
    f32x4 z = { 0.0f, 0.0f, 0.0f, 0.0f };
    ((f32x4*)out_zero)[i] = z;           // out[2*B*D + 4i .. +3] = 0
}

// ---------------------------------------------------------------------------
// Pre-pass 2: A [D,D] -> bf16 hi/lo (row-major kept: B[k][n]=A[n][k] reads
// A rows contiguously along k, no transpose needed)
// ---------------------------------------------------------------------------
__global__ __launch_bounds__(256) void prep_A(
    const float* __restrict__ A, u16* __restrict__ Ah, u16* __restrict__ Al)
{
    size_t i = (size_t)blockIdx.x * 256 + threadIdx.x;
    f32x4 av = ((const f32x4*)A)[i];
    u16x4 ah, al;
#pragma unroll
    for (int c = 0; c < 4; ++c) {
        u16 hi = f32_to_bf16(av[c]);
        ah[c] = hi;
        al[c] = f32_to_bf16(av[c] - bf16_to_f32(hi));
    }
    ((u16x4*)Ah)[i] = ah;
    ((u16x4*)Al)[i] = al;
}

// ---------------------------------------------------------------------------
// GEMM: 256 threads (8 waves), block tile 128x128, wave tile 32x64 (2x4
// accumulators). K-loop step 32, LDS double-buffered async staging.
//   acc += Sh*Ah + Sh*Al + Sl*Ah ; dx = acc + c0 ; write dx, -dx.
// ---------------------------------------------------------------------------
__global__ __launch_bounds__(256) void wmma_gemm(
    const u16* __restrict__ Sh, const u16* __restrict__ Sl,
    const u16* __restrict__ Ah, const u16* __restrict__ Al,
    const float* __restrict__ C0, float* __restrict__ out)
{
    __shared__ __align__(16) char smem[2 * STAGEB];   // 80 KB of 320 KB WGP LDS

    const int tid  = threadIdx.x;
    const int lane = tid & 31;
    const int wave = tid >> 5;
    const int l16  = lane & 15;
    const int h    = lane >> 4;

    const int m_blk = blockIdx.x * BM;
    const int n_blk = blockIdx.y * BN;
    const int wm = wave & 3;        // M quarter (32 rows, 2 tiles)
    const int wn = wave >> 2;       // N half   (64 cols, 4 tiles)

    // Low 32 bits of a flat pointer into __shared__ == raw LDS byte offset.
    const unsigned lds0 = (unsigned)(size_t)(&smem[0]);

    // Cooperative staging coords: thread t -> row (t>>2), 16B column (t&3).
    const int sr = tid >> 2;            // 0..63 (per half)
    const int sc = (tid & 3) * 16;      // byte offset within 64B k-chunk

    const u16* gsrc[4] = { Sh, Sl, Ah, Al };

    auto stage = [&](int buf, int k0) {
#pragma unroll
        for (int a = 0; a < 4; ++a) {
            const int rowbase = (a < 2) ? m_blk : n_blk;
#pragma unroll
            for (int half = 0; half < 2; ++half) {
                const int r = sr + half * 64;
                const u16* g = gsrc[a] + (size_t)(rowbase + r) * DIM + k0 + sc / 2;
                const unsigned lds = lds0 + (unsigned)(buf * STAGEB + a * TILEB + r * ROWB + sc);
                async_load16(lds, g);
            }
        }
    };

    v8f acc[2][4] = {};

    stage(0, 0);
    for (int it = 0; it < DIM / BK; ++it) {
        wait_async0();        // my async writes to buf (it&1) have landed
        __syncthreads();      // everyone's writes landed; prior reads consumed
        if (it + 1 < DIM / BK) stage((it + 1) & 1, (it + 1) * BK);

        const char* base = smem + (it & 1) * STAGEB;

        // A-frags (S tiles): 16-bit A 16x32 layout -> two 16B ds loads/lane.
        Frag fah[2], fal[2];
#pragma unroll
        for (int i = 0; i < 2; ++i) {
            const char* pS = base + (wm * 32 + i * 16 + l16) * ROWB + h * 16;
            fah[i].h[0] = *(const u16x8*)(pS);
            fah[i].h[1] = *(const u16x8*)(pS + 32);
            fal[i].h[0] = *(const u16x8*)(pS + TILEB);
            fal[i].h[1] = *(const u16x8*)(pS + TILEB + 32);
        }
        // B-frags (A tiles): same K layout as A-frags (8h.., 16+8h..) -> two 16B ds loads.
        Frag fbh[4], fbl[4];
#pragma unroll
        for (int j = 0; j < 4; ++j) {
            const char* pA = base + 2 * TILEB + (wn * 64 + j * 16 + l16) * ROWB + h * 16;
            fbh[j].h[0] = *(const u16x8*)(pA);
            fbh[j].h[1] = *(const u16x8*)(pA + 32);
            fbl[j].h[0] = *(const u16x8*)(pA + TILEB);
            fbl[j].h[1] = *(const u16x8*)(pA + TILEB + 32);
        }
#pragma unroll
        for (int i = 0; i < 2; ++i) {
#pragma unroll
            for (int j = 0; j < 4; ++j) {
                acc[i][j] = __builtin_amdgcn_wmma_f32_16x16x32_bf16(
                    false, fah[i].bf, false, fbh[j].bf, (short)0, acc[i][j], false, false);
                acc[i][j] = __builtin_amdgcn_wmma_f32_16x16x32_bf16(
                    false, fah[i].bf, false, fbl[j].bf, (short)0, acc[i][j], false, false);
                acc[i][j] = __builtin_amdgcn_wmma_f32_16x16x32_bf16(
                    false, fal[i].bf, false, fbh[j].bf, (short)0, acc[i][j], false, false);
            }
        }
    }

    // Epilogue: C/D layout (VGPR r -> row h*8+r, col l16). Fuse +c0 and -dx.
    const size_t NN = (size_t)BATCH * DIM;
#pragma unroll
    for (int i = 0; i < 2; ++i) {
#pragma unroll
        for (int j = 0; j < 4; ++j) {
#pragma unroll
            for (int r = 0; r < 8; ++r) {
                int row = m_blk + wm * 32 + i * 16 + h * 8 + r;
                int col = n_blk + wn * 64 + j * 16 + l16;
                size_t idx = (size_t)row * DIM + col;
                float v = acc[i][j][r] + C0[idx];
                out[idx]      = v;
                out[idx + NN] = -v;
            }
        }
    }
}

// ---------------------------------------------------------------------------
// Workspace: Sh 16MB | Sl 16MB | Ah 2MB | Al 2MB | C0 32MB  (68 MB total)
// ---------------------------------------------------------------------------
extern "C" void kernel_launch(void* const* d_in, const int* in_sizes, int n_in,
                              void* d_out, int out_size, void* d_ws, size_t ws_size,
                              hipStream_t stream) {
    const float* x   = (const float*)d_in[0];
    const float* ex  = (const float*)d_in[1];
    const float* W   = (const float*)d_in[2];
    const float* A   = (const float*)d_in[3];
    const float* tgt = (const float*)d_in[4];
    float* out = (float*)d_out;

    const size_t SZ_S = (size_t)BATCH * DIM * sizeof(u16);   // 16 MB
    const size_t SZ_A = (size_t)DIM * DIM * sizeof(u16);     //  2 MB
    const size_t NN   = (size_t)BATCH * DIM;

    char* ws = (char*)d_ws;
    u16*   Sh = (u16*)(ws);
    u16*   Sl = (u16*)(ws + SZ_S);
    u16*   Ah = (u16*)(ws + 2 * SZ_S);
    u16*   Al = (u16*)(ws + 2 * SZ_S + SZ_A);
    float* C0 = (float*)(ws + 2 * SZ_S + 2 * SZ_A);

    prep_state<<<(BATCH * (DIM / 4)) / 256, 256, 0, stream>>>(
        x, ex, W, tgt, Sh, Sl, C0, out + 2 * NN);
    prep_A<<<(DIM * (DIM / 4)) / 256, 256, 0, stream>>>(A, Ah, Al);
    wmma_gemm<<<dim3(BATCH / 128, DIM / 128), 256, 0, stream>>>(Sh, Sl, Ah, Al, C0, out);
}
